// EALSTM_18494129177333
// MI455X (gfx1250) — matrix operand order repaired and checked by reading, not compiled, not run on hardware
//
#include <hip/hip_runtime.h>
#include <hip/hip_bf16.h>

// Problem constants
#define B_  256
#define T_  365
#define D_  32
#define S_  27
#define U_  256
#define N3_ 768            // 3*U
#define NTILES 48          // N3_/16
#define KCH 9              // packed k-chunks of 32: [0]=input kernel (D=32), [1..8]=rec_kernel (U=256)

typedef __attribute__((ext_vector_type(16))) __bf16 v16bf;
typedef __attribute__((ext_vector_type(8)))  float  v8f;
typedef __attribute__((ext_vector_type(4)))  unsigned int v4u;   // native vec for NT builtins

union Frag16 { v16bf v; __bf16 e[16]; uint4 q[2]; };

__device__ __forceinline__ float fast_sigmoid(float x) {
    return 1.0f / (1.0f + __expf(-x));
}
__device__ __forceinline__ float fast_tanh(float x) {
    return 1.0f - 2.0f / (1.0f + __expf(2.0f * x));   // tanh(x) = 1 - 2/(exp(2x)+1)
}

// Unpack 8 bf16 (packed in a uint4) to f32 via exponent-preserving shifts.
__device__ __forceinline__ void bf16x8_to_v8f(const uint4 q, v8f& a) {
    const unsigned int u[4] = {q.x, q.y, q.z, q.w};
#pragma unroll
    for (int i = 0; i < 4; ++i) {
        a[2 * i]     = __uint_as_float(u[i] << 16);
        a[2 * i + 1] = __uint_as_float(u[i] & 0xffff0000u);
    }
}

// -----------------------------------------------------------------------------
// Prep: pack [kernel (32x768); rec_kernel (256x768)] as bf16 WMMA B-fragments.
// Fragment (kc, tn): 32 lanes x 16 bf16 lane-contiguous (32 B per lane).
// B element j of lane l -> (K = (j<8 ? (l>>4)*8 + j : 16 + (l>>4)*8 + j-8), n = tn*16 + (l&15)).
// -----------------------------------------------------------------------------
__global__ void ealstm_prep_weights(const float* __restrict__ kern,
                                    const float* __restrict__ reck,
                                    __bf16* __restrict__ wpack)
{
    const int e = blockIdx.x * 256 + threadIdx.x;
    if (e >= KCH * NTILES * 32 * 16) return;
    const int j    = e & 15;
    const int lane = (e >> 4) & 31;
    const int rem  = e >> 9;            // kc*48 + tn
    const int tn   = rem % NTILES;
    const int kc   = rem / NTILES;
    const int n    = tn * 16 + (lane & 15);
    const int Kl   = ((j >> 3) << 4) + ((lane >> 4) << 3) + (j & 7);
    const float v  = (kc == 0) ? kern[Kl * N3_ + n]
                               : reck[((kc - 1) * 32 + Kl) * N3_ + n];
    wpack[e] = (__bf16)v;
}

// -----------------------------------------------------------------------------
// Input projection: x_pre[b,t,:] = x[b,t,:] @ kernel + bias, stored as bf16 in
// C-fragment layout: frag (wg, tn, t) = 32 lanes x 16 B; lane holds its 8 values.
// Grid (16, 365) x 512 threads -> fully parallel WMMA GEMM across the chip.
// x_pre is a stream-once 143 MB buffer: store non-temporal to spare L2 for the
// packed weights that every xproj block re-reads.
// -----------------------------------------------------------------------------
__global__ void __launch_bounds__(512, 1)
ealstm_xproj(const float* __restrict__ x,        // (B,T,D) f32
             const float* __restrict__ bias,     // (3U,)
             const __bf16* __restrict__ wpack,   // packed B fragments (kc=0 used)
             __bf16* __restrict__ xpre)          // fragment-layout output
{
    const int tid  = threadIdx.x;
    const int lane = tid & 31;
    const int uw   = tid >> 5;           // 0..15
    const int nl   = lane & 15;
    const int hi   = lane >> 4;
    const int wg   = blockIdx.x;         // batch chunk
    const int t    = blockIdx.y;         // timestep

    // A = x_t tile (16 x 32), f32 -> bf16
    Frag16 A;
    {
        const float* xr = x + ((size_t)(wg * 16 + nl) * T_ + t) * D_;
        const int d0 = hi ? 8 : 0;
        const int d1 = hi ? 24 : 16;
        float4 f0 = *(const float4*)(xr + d0);
        float4 f1 = *(const float4*)(xr + d0 + 4);
        float4 f2 = *(const float4*)(xr + d1);
        float4 f3 = *(const float4*)(xr + d1 + 4);
        A.e[0]  = (__bf16)f0.x; A.e[1]  = (__bf16)f0.y;
        A.e[2]  = (__bf16)f0.z; A.e[3]  = (__bf16)f0.w;
        A.e[4]  = (__bf16)f1.x; A.e[5]  = (__bf16)f1.y;
        A.e[6]  = (__bf16)f1.z; A.e[7]  = (__bf16)f1.w;
        A.e[8]  = (__bf16)f2.x; A.e[9]  = (__bf16)f2.y;
        A.e[10] = (__bf16)f2.z; A.e[11] = (__bf16)f2.w;
        A.e[12] = (__bf16)f3.x; A.e[13] = (__bf16)f3.y;
        A.e[14] = (__bf16)f3.z; A.e[15] = (__bf16)f3.w;
    }

    const uint4* wq = (const uint4*)wpack;
    const size_t lo = (size_t)lane * 2;

#pragma unroll
    for (int g = 0; g < 3; ++g) {
        const int tn = uw + g * 16;
        Frag16 Bf;
        const uint4* p = wq + (size_t)tn * 64 + lo;     // kc = 0 fragments
        Bf.q[0] = p[0]; Bf.q[1] = p[1];

        const float bv = bias[tn * 16 + nl];
        v8f acc;
#pragma unroll
        for (int r = 0; r < 8; ++r) acc[r] = bv;
        acc = __builtin_amdgcn_wmma_f32_16x16x32_bf16(false, A.v, false, Bf.v,
                                                      (short)0, acc, false, false);
        union { v4u q; __bf16 e[8]; } o;
#pragma unroll
        for (int r = 0; r < 8; ++r) o.e[r] = (__bf16)acc[r];
        // fragment = 512 B = 32 x 16B; lane stores its own 16 B (non-temporal)
        __builtin_nontemporal_store(
            o.q, (v4u*)xpre + ((size_t)(wg * NTILES + tn) * T_ + t) * 32 + lane);
    }
}

// -----------------------------------------------------------------------------
// Recurrent kernel: 16 blocks (16 batch rows) x 512 threads (16 waves).
// All 24 recurrent B-fragments live in VGPRs (192) for all 365 steps.
// Accumulators initialized from precomputed x_pre fragments (1 b128/lane each);
// the next step's fragments are software-prefetched (global_prefetch_b8) so the
// serial loop never waits on HBM. h round-trips through LDS as bf16.
// -----------------------------------------------------------------------------
__global__ void __launch_bounds__(512, 1)
ealstm_recurrent(const float* __restrict__ stat,    // (B,S)
                 const float* __restrict__ wstat,   // (S,U)
                 const float* __restrict__ bstat,   // (U,)
                 const __bf16* __restrict__ wpack,  // packed B fragments (kc 1..8)
                 const __bf16* __restrict__ xpre,   // precomputed x@kernel+bias
                 float* __restrict__ out)           // (B,U)
{
    __shared__ __align__(16) __bf16 hS[16 * U_];

    const int tid  = threadIdx.x;
    const int lane = tid & 31;
    const int uw   = tid >> 5;
    const int nl   = lane & 15;
    const int hi   = lane >> 4;
    const int mb   = hi ? 8 : 0;
    const int wg   = blockIdx.x;
    const int ncol = uw * 16 + nl;

    ((uint4*)hS)[tid] = make_uint4(0u, 0u, 0u, 0u);   // h0 = 0

    // Static input gate (time-invariant) + cell state in registers.
    float ig[8], cS[8];
#pragma unroll
    for (int r = 0; r < 8; ++r) {
        const int m = mb + r;
        float a = bstat[ncol];
        const float* srow = stat + (size_t)(wg * 16 + m) * S_;
        for (int s = 0; s < S_; ++s) a += srow[s] * wstat[s * U_ + ncol];
        ig[r] = fast_sigmoid(a);
        cS[r] = 0.0f;
    }
    __syncthreads();

    // Preload all 24 recurrent B fragments into registers (kc 1..8 of wpack).
    Frag16 Bw[8][3];
    {
        const uint4* wq = (const uint4*)wpack;
        const size_t lo = (size_t)lane * 2;
#pragma unroll
        for (int kc = 0; kc < 8; ++kc)
#pragma unroll
            for (int g = 0; g < 3; ++g) {
                const uint4* p = wq + ((size_t)((kc + 1) * NTILES) + uw + g * 16) * 64 + lo;
                Bw[kc][g].q[0] = p[0];
                Bw[kc][g].q[1] = p[1];
            }
    }

    const uint4* xq = (const uint4*)xpre;
    const size_t xbF = ((size_t)(wg * NTILES + uw)      * T_) * 32 + lane;
    const size_t xbC = ((size_t)(wg * NTILES + uw + 16) * T_) * 32 + lane;
    const size_t xbO = ((size_t)(wg * NTILES + uw + 32) * T_) * 32 + lane;

    for (int t = 0; t < T_; ++t) {
        const size_t to = (size_t)t * 32;

        // Accumulators initialized with x_t @ kernel + bias (precomputed).
        v8f aF, aC, aO;
        bf16x8_to_v8f(xq[xbF + to], aF);
        bf16x8_to_v8f(xq[xbC + to], aC);
        bf16x8_to_v8f(xq[xbO + to], aO);

        // Prefetch next step's fragments while this step's WMMAs run.
        if (t + 1 < T_) {
            __builtin_prefetch(xq + xbF + to + 32, 0, 1);
            __builtin_prefetch(xq + xbC + to + 32, 0, 1);
            __builtin_prefetch(xq + xbO + to + 32, 0, 1);
        }

#pragma unroll
        for (int kc = 0; kc < 8; ++kc) {
            Frag16 A;       // h tile (16 x 32 K-slice) from LDS
            const __bf16* hr = hS + nl * U_ + kc * 32;
            A.q[0] = *(const uint4*)(hr + (hi ? 8 : 0));
            A.q[1] = *(const uint4*)(hr + (hi ? 24 : 16));
            aF = __builtin_amdgcn_wmma_f32_16x16x32_bf16(false, A.v, false, Bw[kc][0].v,
                                                         (short)0, aF, false, false);
            aC = __builtin_amdgcn_wmma_f32_16x16x32_bf16(false, A.v, false, Bw[kc][1].v,
                                                         (short)0, aC, false, false);
            aO = __builtin_amdgcn_wmma_f32_16x16x32_bf16(false, A.v, false, Bw[kc][2].v,
                                                         (short)0, aO, false, false);
        }

        __syncthreads();    // all waves done reading hS

#pragma unroll
        for (int r = 0; r < 8; ++r) {
            const float f  = fast_sigmoid(aF[r]);
            const float ct = fast_tanh(aC[r]);
            const float o  = fast_sigmoid(aO[r]);
            cS[r] = f * cS[r] + ig[r] * ct;
            hS[(mb + r) * U_ + ncol] = (__bf16)(o * fast_tanh(cS[r]));
        }
        __syncthreads();    // new h visible
    }

    // Final hidden state (read back own bf16 h, widen to f32).
#pragma unroll
    for (int r = 0; r < 8; ++r)
        out[(size_t)(wg * 16 + mb + r) * U_ + ncol] = (float)hS[(mb + r) * U_ + ncol];
}

// -----------------------------------------------------------------------------
extern "C" void kernel_launch(void* const* d_in, const int* in_sizes, int n_in,
                              void* d_out, int out_size, void* d_ws, size_t ws_size,
                              hipStream_t stream)
{
    const float* inputs  = (const float*)d_in[0];  // (B,T,D)
    const float* statics = (const float*)d_in[1];  // (B,S)
    const float* kern    = (const float*)d_in[2];  // (D,3U)
    const float* reck    = (const float*)d_in[3];  // (U,3U)
    const float* bias    = (const float*)d_in[4];  // (3U,)
    const float* wstat   = (const float*)d_in[5];  // (S,U)
    const float* bstat   = (const float*)d_in[6];  // (U,)
    float* out = (float*)d_out;                    // (B,U)

    // Workspace layout: [0, 432KB) packed weights; [512KB, ...) x_pre fragments.
    __bf16* wpack = (__bf16*)d_ws;
    __bf16* xpre  = (__bf16*)((char*)d_ws + (512u << 10));

    // Pack weights: 9*48*32*16 = 221184 bf16
    ealstm_prep_weights<<<864, 256, 0, stream>>>(kern, reck, wpack);

    // Input projection for all (b,t): grid (16, 365), WMMA per wave
    ealstm_xproj<<<dim3(B_ / 16, T_), 512, 0, stream>>>(inputs, bias, wpack, xpre);

    // Sequential recurrence: 16 batch chunks x 16 waves
    ealstm_recurrent<<<B_ / 16, 512, 0, stream>>>(statics, wstat, bstat, wpack,
                                                  xpre, out);
}
